// VDPWIModel_74423193305823
// MI455X (gfx1250) — matrix operand order repaired and checked by reading, not compiled, not run on hardware
//
#include <hip/hip_runtime.h>
#include <math.h>

typedef __attribute__((ext_vector_type(16))) _Float16 v16h;
typedef __attribute__((ext_vector_type(8)))  float    v8f;

#define T48 48
#define HID 250
#define G4H 1000
#define GLD 1008   // padded gate width (63 * 16)
#define EMBD 300
#define EKP 320    // padded K for embed gemm (10 k-tiles)

// ---- fragment swizzle helpers (match CDNA5 WMMA VGPR layouts) ----
// A (16x32 f16): lane r=lane&15 -> row m; hi=lane>>4; element e of v16h maps to
// k = (vv<4 ? 2vv : 16+2(vv-4)) + hi*8 + (e&1), vv=e>>1.   [ISA 7.12.2]
// B (32x16 f16): same k order as A: k = (e<8 ? e : 16+(e-8)) + hi*8; n = lane&15.
__device__ __forceinline__ int a_k_of(int kt, int lane, int e) {
  int hi = lane >> 4, vv = e >> 1;
  int kk = ((vv < 4) ? (vv << 1) : (16 + ((vv - 4) << 1))) + (hi << 3) + (e & 1);
  return (kt << 5) + kk;
}

// Swizzled A storage: A[((mtile*K32 + kt)*32 + lane)*16 + e]
// Swizzled B storage: B[((kt*NT + ntile)*32 + lane)*16 + e]

// ---------------- packing kernels ----------------

__global__ void k_embed_pack(const float* __restrict__ emb, const int* __restrict__ ids,
                             _Float16* __restrict__ A /* swizzled 48x320 */) {
  int idx = blockIdx.x * blockDim.x + threadIdx.x;
  if (idx >= T48 * EKP) return;
  int e = idx & 15, lane = (idx >> 4) & 31, rest = idx >> 9;
  int kt = rest % (EKP / 32), mtile = rest / (EKP / 32);
  int m = (mtile << 4) + (lane & 15);               // m < 48 always
  int k = a_k_of(kt, lane, e);
  float v = (k < EMBD) ? emb[(size_t)ids[m] * EMBD + k] : 0.0f;
  A[idx] = (_Float16)v;
}

// generic weight pack: source w is N x K row-major (value at w[n*K + k])
__global__ void k_packB(const float* __restrict__ w, _Float16* __restrict__ B,
                        int K, int N, int Kp, int Np) {
  int idx = blockIdx.x * blockDim.x + threadIdx.x;
  if (idx >= Kp * Np) return;
  int ntn = Np >> 4;
  int e = idx & 15, lane = (idx >> 4) & 31, rest = idx >> 9;
  int kt = rest / ntn, ntile = rest % ntn;
  int k = (kt << 5) + ((lane >> 4) << 3) + ((e >> 3) << 4) + (e & 7);
  int n = (ntile << 4) + (lane & 15);
  float v = (k < K && n < N) ? w[(size_t)n * K + k] : 0.0f;
  B[idx] = (_Float16)v;
}

// ---------------- WMMA GEMM over swizzled operands ----------------
// C(MxN,f32 row-major) = A * B ; M%16==0, N%16==0, K%32==0.
// One 16x16 C tile per wave; inner loop = 2x b128 (A) + 2x b128 (B) + 1 wmma.

__global__ void k_gemm_f16f32(const _Float16* __restrict__ A,
                              const _Float16* __restrict__ B,
                              float* __restrict__ C, int M, int N, int K) {
  int lane = threadIdx.x;                        // 0..31
  int tile = blockIdx.x * blockDim.y + threadIdx.y;
  int nt = N >> 4;
  int K32 = K >> 5;
  if (tile >= (M >> 4) * nt) return;             // wave-uniform exit
  int mtile = tile / nt, ntile = tile % nt;

  const v16h* Ap = (const v16h*)A + ((size_t)mtile * K32) * 32 + lane;
  const v16h* Bp = (const v16h*)B + (size_t)ntile * 32 + lane;

  v8f acc = {};
  #pragma unroll 2
  for (int kt = 0; kt < K32; ++kt) {
    v16h a = Ap[(size_t)kt * 32];
    v16h b = Bp[(size_t)kt * nt * 32];
    acc = __builtin_amdgcn_wmma_f32_16x16x32_f16(
        false, a, false, b, (short)0, acc, false, false);
  }
  int r = lane & 15, hi = lane >> 4;
  float* Crow = C + (size_t)((mtile << 4) + hi * 8) * N + ((ntile << 4) + r);
  #pragma unroll
  for (int rr = 0; rr < 8; ++rr) Crow[(size_t)rr * N] = acc[rr];
}

// ---------------- LSTM recurrence (4 independent runs) ----------------

__global__ void k_lstm(const float* __restrict__ xW /* 4 x 48 x 1008 */,
                       const float* __restrict__ w_hh_f, const float* __restrict__ b_f,
                       const float* __restrict__ w_hh_b, const float* __restrict__ b_b,
                       float* __restrict__ hout /* 4 x 48 x 250 */) {
  int combo = blockIdx.x;                  // 0:(f,s1) 1:(f,s2) 2:(b,s1) 3:(b,s2)
  bool rev = combo >= 2;
  const float* whh = rev ? w_hh_b : w_hh_f;
  const float* bia = rev ? b_b : b_f;
  const float* xw  = xW + (size_t)combo * T48 * GLD;
  float* hs        = hout + (size_t)combo * T48 * HID;

  __shared__ float h[HID], c[HID], g[G4H];
  int tid = threadIdx.x;
  if (tid < HID) { h[tid] = 0.0f; c[tid] = 0.0f; }
  __syncthreads();

  for (int step = 0; step < T48; ++step) {
    int t = rev ? (T48 - 1 - step) : step;
    for (int gi = tid; gi < G4H; gi += 256) {
      const float* wr = whh + (size_t)gi * HID;
      float acc = bia[gi] + xw[(size_t)t * GLD + gi];
      #pragma unroll 5
      for (int k = 0; k < HID; ++k) acc = fmaf(h[k], wr[k], acc);
      g[gi] = acc;
    }
    __syncthreads();
    if (tid < HID) {
      float ig = 1.0f / (1.0f + expf(-g[tid]));
      float fg = 1.0f / (1.0f + expf(-g[HID + tid]));
      float gg = tanhf(g[2 * HID + tid]);
      float og = 1.0f / (1.0f + expf(-g[3 * HID + tid]));
      float cn = fg * c[tid] + ig * gg;
      float hn = og * tanhf(cn);
      c[tid] = cn; h[tid] = hn;
      hs[(size_t)t * HID + tid] = hn;
    }
    __syncthreads();
  }
}

// ---------------- norms / similarity cube ----------------

__global__ void k_norms(const float* __restrict__ hbuf, float* __restrict__ norms /*4x96*/) {
  int row = blockIdx.x * blockDim.x + threadIdx.x;
  if (row >= 96) return;
  int seq = row / 48, r = row % 48;
  const float* hf = hbuf + ((size_t)seq * T48 + r) * HID;        // combo 0/1
  const float* hb = hbuf + ((size_t)(2 + seq) * T48 + r) * HID;  // combo 2/3
  float sf = 0.f, sb = 0.f, sa = 0.f;
  for (int k = 0; k < HID; ++k) {
    float a = hf[k], b = hb[k], s = a + b;
    sf += a * a; sb += b * b; sa += s * s;
  }
  norms[row]        = sqrtf(sf);        // nf
  norms[96 + row]   = sqrtf(sb);        // nb
  norms[192 + row]  = sqrtf(sf + sb);   // ncat
  norms[288 + row]  = sqrtf(sa);        // nadd
}

__device__ __forceinline__ void sim_write(float* sc, int p, int idx,
                                          float dot, float na, float nb) {
  sc[(size_t)p * 2304 + idx]       = dot;
  sc[(size_t)(p + 1) * 2304 + idx] = dot / (na * nb + 1e-8f);
  float d2 = fmaxf(na * na + nb * nb - 2.0f * dot, 1e-12f);
  sc[(size_t)(p + 2) * 2304 + idx] = sqrtf(d2);
}

__global__ void k_simcube(const float* __restrict__ hbuf, const float* __restrict__ norms,
                          float* __restrict__ sc /* 13 x 2304 */) {
  int idx = blockIdx.x * blockDim.x + threadIdx.x;
  if (idx >= 2304) return;
  int i = idx / 48, j = idx % 48;
  const float* h1f = hbuf + ((size_t)0 * T48 + i) * HID;
  const float* h2f = hbuf + ((size_t)1 * T48 + j) * HID;
  const float* h1b = hbuf + ((size_t)2 * T48 + i) * HID;
  const float* h2b = hbuf + ((size_t)3 * T48 + j) * HID;
  float ff = 0.f, bb = 0.f, fb = 0.f, bf = 0.f;
  for (int k = 0; k < HID; ++k) {
    float a = h1f[k], b = h1b[k], cc = h2f[k], d = h2b[k];
    ff += a * cc; bb += b * d; fb += a * d; bf += b * cc;
  }
  sim_write(sc, 0, idx, ff + bb,           norms[192 + i], norms[192 + 48 + j]); // cat
  sim_write(sc, 3, idx, ff,                norms[i],       norms[48 + j]);       // fwd
  sim_write(sc, 6, idx, bb,                norms[96 + i],  norms[96 + 48 + j]);  // bwd
  sim_write(sc, 9, idx, ff + fb + bf + bb, norms[288 + i], norms[288 + 48 + j]); // add
  sc[(size_t)12 * 2304 + idx] = 0.0f;                                            // pad
}

// ---------------- focus cube: greedy top-96 on planes 10, 11 ----------------

__global__ void k_focus(const float* __restrict__ sc, float* __restrict__ mask /*2304*/) {
  __shared__ float buf[2304];
  __shared__ float rv[256];
  __shared__ int   ri[256];
  __shared__ int s1[48], s2[48];
  __shared__ unsigned char selm[2304];
  int tid = threadIdx.x;
  for (int e = tid; e < 2304; e += 256) selm[e] = 0;

  for (int pl = 0; pl < 2; ++pl) {
    const float* plane = sc + (size_t)(10 + pl) * 2304;
    for (int e = tid; e < 2304; e += 256) buf[e] = plane[e];
    if (tid < 48) { s1[tid] = 0; s2[tid] = 0; }
    __syncthreads();
    for (int it = 0; it < 96; ++it) {
      float bv = -3.0e38f; int bi = 1 << 30;
      for (int e = tid; e < 2304; e += 256) {       // strictly > keeps lowest idx
        float v = buf[e];
        if (v > bv) { bv = v; bi = e; }
      }
      rv[tid] = bv; ri[tid] = bi;
      __syncthreads();
      for (int s = 128; s > 0; s >>= 1) {
        if (tid < s) {
          if (rv[tid + s] > rv[tid] ||
              (rv[tid + s] == rv[tid] && ri[tid + s] < ri[tid])) {
            rv[tid] = rv[tid + s]; ri[tid] = ri[tid + s];
          }
        }
        __syncthreads();
      }
      if (tid == 0) {
        int idx = ri[0]; int p1 = idx / 48, p2 = idx % 48;
        if (s1[p1] + s2[p2] == 0) { s1[p1] = 1; s2[p2] = 1; selm[idx] = 1; }
        buf[idx] = -3.0e38f;
      }
      __syncthreads();
    }
    __syncthreads();
  }
  for (int e = tid; e < 2304; e += 256) mask[e] = selm[e] ? 1.0f : 0.1f;
}

__global__ void k_apply_focus(const float* __restrict__ sc, const float* __restrict__ mask,
                              float* __restrict__ fc) {
  int idx = blockIdx.x * blockDim.x + threadIdx.x;
  if (idx >= 13 * 2304) return;
  int p = idx / 2304, e = idx % 2304;
  float m = (p == 12) ? 1.0f : mask[e];
  fc[idx] = sc[idx] * m;
}

// ---------------- conv net pieces (implicit GEMM, swizzled A) ----------------

__global__ void k_im2col(const float* __restrict__ in /* C x HW x HW */,
                         _Float16* __restrict__ A, int C, int HW, int K, int Mp, int Kp) {
  int idx = blockIdx.x * blockDim.x + threadIdx.x;
  if (idx >= Mp * Kp) return;
  int K32 = Kp >> 5;
  int e = idx & 15, lane = (idx >> 4) & 31, rest = idx >> 9;
  int kt = rest % K32, mtile = rest / K32;
  int m = (mtile << 4) + (lane & 15);
  int k = a_k_of(kt, lane, e);
  float v = 0.0f;
  int M = HW * HW;
  if (m < M && k < K) {
    int ch = k / 9, rem = k % 9, dy = rem / 3, dx = rem % 3;
    int y = m / HW, x = m % HW;
    int iy = y + dy - 1, ix = x + dx - 1;
    if (iy >= 0 && iy < HW && ix >= 0 && ix < HW)
      v = in[((size_t)ch * HW + iy) * HW + ix];
  }
  A[idx] = (_Float16)v;
}

__global__ void k_bias_relu_pool(const float* __restrict__ Cmat /* (HW*HW) x Np */,
                                 const float* __restrict__ bias,
                                 float* __restrict__ out /* Cout x oHW x oHW */,
                                 int HW, int Np, int Cout, int pk, int ps, int oHW) {
  int idx = blockIdx.x * blockDim.x + threadIdx.x;
  if (idx >= Cout * oHW * oHW) return;
  int n = idx / (oHW * oHW);
  int rem = idx % (oHW * oHW);
  int py = rem / oHW, px = rem % oHW;
  float mx = -3.0e38f;
  for (int dy = 0; dy < pk; ++dy)
    for (int dx = 0; dx < pk; ++dx) {
      int y = py * ps + dy, x = px * ps + dx;
      float v = Cmat[((size_t)y * HW + x) * Np + n] + bias[n];
      mx = fmaxf(mx, v);
    }
  out[idx] = fmaxf(mx, 0.0f);
}

// ---------------- head ----------------

__global__ void k_head(const float* __restrict__ x128, const float* __restrict__ dnn_w,
                       const float* __restrict__ dnn_b, const float* __restrict__ out_w,
                       const float* __restrict__ out_b, float* __restrict__ out) {
  __shared__ float x[128], y[128], z[5], lse[1];
  int tid = threadIdx.x;  // 128 threads
  x[tid] = x128[tid];
  __syncthreads();
  {
    float acc = dnn_b[tid];
    const float* wr = dnn_w + (size_t)tid * 128;
    for (int k = 0; k < 128; ++k) acc = fmaf(x[k], wr[k], acc);
    y[tid] = fmaxf(acc, 0.0f);
  }
  __syncthreads();
  if (tid < 5) {
    float a = out_b[tid];
    const float* orow = out_w + (size_t)tid * 128;
    for (int k = 0; k < 128; ++k) a = fmaf(y[k], orow[k], a);
    z[tid] = a;
  }
  __syncthreads();
  if (tid == 0) {
    float mx = z[0];
    for (int i = 1; i < 5; ++i) mx = fmaxf(mx, z[i]);
    float s = 0.0f;
    for (int i = 0; i < 5; ++i) s += expf(z[i] - mx);
    lse[0] = mx + logf(s);
  }
  __syncthreads();
  if (tid < 5) out[tid] = z[tid] - lse[0];
}

// ---------------- launcher ----------------

static inline int cdiv(int a, int b) { return (a + b - 1) / b; }

extern "C" void kernel_launch(void* const* d_in, const int* in_sizes, int n_in,
                              void* d_out, int out_size, void* d_ws, size_t ws_size,
                              hipStream_t stream) {
  (void)in_sizes; (void)n_in; (void)out_size; (void)ws_size;
  const int*   x1     = (const int*)  d_in[0];
  const int*   x2     = (const int*)  d_in[1];
  const float* emb    = (const float*)d_in[2];
  const float* w_ih_f = (const float*)d_in[3];
  const float* w_hh_f = (const float*)d_in[4];
  const float* b_f    = (const float*)d_in[5];
  const float* w_ih_b = (const float*)d_in[6];
  const float* w_hh_b = (const float*)d_in[7];
  const float* b_b    = (const float*)d_in[8];
  const float* cw[5]  = {(const float*)d_in[9],  (const float*)d_in[11],
                         (const float*)d_in[13], (const float*)d_in[15],
                         (const float*)d_in[17]};
  const float* cb[5]  = {(const float*)d_in[10], (const float*)d_in[12],
                         (const float*)d_in[14], (const float*)d_in[16],
                         (const float*)d_in[18]};
  const float* dnn_w  = (const float*)d_in[19];
  const float* dnn_b  = (const float*)d_in[20];
  const float* out_w  = (const float*)d_in[21];
  const float* out_b  = (const float*)d_in[22];

  // workspace layout (deterministic, 256B aligned)
  char* base = (char*)d_ws;
  size_t off = 0;
  auto take = [&](size_t bytes) {
    size_t o = off;
    off += (bytes + 255) & ~(size_t)255;
    return o;
  };
  _Float16* eA1  = (_Float16*)(base + take((size_t)T48 * EKP * 2));
  _Float16* eA2  = (_Float16*)(base + take((size_t)T48 * EKP * 2));
  _Float16* eBf  = (_Float16*)(base + take((size_t)EKP * GLD * 2));
  _Float16* eBb  = (_Float16*)(base + take((size_t)EKP * GLD * 2));
  float*    xW   = (float*)   (base + take((size_t)4 * T48 * GLD * 4));
  float*    hbuf = (float*)   (base + take((size_t)4 * T48 * HID * 4));
  float*    nrm  = (float*)   (base + take((size_t)4 * 96 * 4));
  float*    sc   = (float*)   (base + take((size_t)13 * 2304 * 4));
  float*    mask = (float*)   (base + take((size_t)2304 * 4));
  float*    fc   = (float*)   (base + take((size_t)13 * 2304 * 4));
  _Float16* gA   = (_Float16*)(base + take((size_t)576 * 1152 * 2));   // max im2col
  _Float16* gB   = (_Float16*)(base + take((size_t)1728 * 192 * 2));   // max weights
  float*    gC   = (float*)   (base + take((size_t)2304 * 128 * 4));   // max gemm out
  float*    act0 = (float*)   (base + take((size_t)128 * 24 * 24 * 4));
  float*    act1 = (float*)   (base + take((size_t)128 * 24 * 24 * 4));

  // --- embedding gather + input GEMMs (WMMA) ---
  k_embed_pack<<<cdiv(T48 * EKP, 256), 256, 0, stream>>>(emb, x1, eA1);
  k_embed_pack<<<cdiv(T48 * EKP, 256), 256, 0, stream>>>(emb, x2, eA2);
  k_packB<<<cdiv(EKP * GLD, 256), 256, 0, stream>>>(w_ih_f, eBf, EMBD, G4H, EKP, GLD);
  k_packB<<<cdiv(EKP * GLD, 256), 256, 0, stream>>>(w_ih_b, eBb, EMBD, G4H, EKP, GLD);
  {
    int tiles = (T48 / 16) * (GLD / 16);  // 3 * 63
    dim3 blk(32, 8);
    k_gemm_f16f32<<<cdiv(tiles, 8), blk, 0, stream>>>(eA1, eBf, xW + (size_t)0 * T48 * GLD, T48, GLD, EKP);
    k_gemm_f16f32<<<cdiv(tiles, 8), blk, 0, stream>>>(eA2, eBf, xW + (size_t)1 * T48 * GLD, T48, GLD, EKP);
    k_gemm_f16f32<<<cdiv(tiles, 8), blk, 0, stream>>>(eA1, eBb, xW + (size_t)2 * T48 * GLD, T48, GLD, EKP);
    k_gemm_f16f32<<<cdiv(tiles, 8), blk, 0, stream>>>(eA2, eBb, xW + (size_t)3 * T48 * GLD, T48, GLD, EKP);
  }

  // --- LSTM recurrences / sim cube / focus ---
  k_lstm<<<4, 256, 0, stream>>>(xW, w_hh_f, b_f, w_hh_b, b_b, hbuf);
  k_norms<<<1, 128, 0, stream>>>(hbuf, nrm);
  k_simcube<<<cdiv(2304, 256), 256, 0, stream>>>(hbuf, nrm, sc);
  k_focus<<<1, 256, 0, stream>>>(sc, mask);
  k_apply_focus<<<cdiv(13 * 2304, 256), 256, 0, stream>>>(sc, mask, fc);

  // --- conv net: im2col + WMMA GEMM per layer ---
  struct Layer { const float* w; const float* b; int Cin, HW, Cout, pk, ps, oHW; };
  Layer Ls[5] = {
    {cw[0], cb[0],  13, 48, 128, 2, 2, 24},
    {cw[1], cb[1], 128, 24, 164, 2, 2, 12},
    {cw[2], cb[2], 164, 12, 192, 2, 2,  6},
    {cw[3], cb[3], 192,  6, 192, 2, 2,  3},
    {cw[4], cb[4], 192,  3, 128, 3, 1,  1},
  };
  const float* cur = fc;
  float* outs[2] = {act0, act1};
  for (int l = 0; l < 5; ++l) {
    const Layer& L = Ls[l];
    int M = L.HW * L.HW, K = L.Cin * 9, N = L.Cout;
    int Mp = (M + 15) & ~15, Kp = (K + 31) & ~31, Np = (N + 15) & ~15;
    k_im2col<<<cdiv(Mp * Kp, 256), 256, 0, stream>>>(cur, gA, L.Cin, L.HW, K, Mp, Kp);
    k_packB<<<cdiv(Kp * Np, 256), 256, 0, stream>>>(L.w, gB, K, N, Kp, Np);
    int tiles = (Mp / 16) * (Np / 16);
    k_gemm_f16f32<<<cdiv(tiles, 8), dim3(32, 8), 0, stream>>>(gA, gB, gC, Mp, Np, Kp);
    float* outp = outs[l & 1];
    k_bias_relu_pool<<<cdiv(N * L.oHW * L.oHW, 256), 256, 0, stream>>>(
        gC, L.b, outp, L.HW, Np, N, L.pk, L.ps, L.oHW);
    cur = outp;
  }

  // --- MLP head + log_softmax ---
  k_head<<<1, 128, 0, stream>>>(cur, dnn_w, dnn_b, out_w, out_b, (float*)d_out);
}
